// Attention_47373489275233
// MI455X (gfx1250) — matrix operand order repaired and checked by reading, not compiled, not run on hardware
//
#include <hip/hip_runtime.h>
#include <hip/hip_bf16.h>
#include <math.h>

// ---------------------------------------------------------------------------
// CDNA5 (gfx1250) attention block: rmsnorm -> QKV gemm -> flash attn -> out gemm
// All GEMM-shaped math on v_wmma_f32_16x16x32_f16 (wave32).
// GEMM waves compute 32x64 C tiles (8 wmma / K-step, B-frags reused x2).
// ---------------------------------------------------------------------------

typedef __attribute__((ext_vector_type(16))) _Float16 v16h;
typedef __attribute__((ext_vector_type(8)))  _Float16 v8h;
typedef __attribute__((ext_vector_type(8)))  float    v8f;

#define DIM 1024
#define HEADS 16
#define DHEAD 64
#define SEQ 2048
#define BATCH 2
#define ROWS (BATCH*SEQ)          // 4096
#define QKV_COLS (3*DIM)          // 3072

__device__ __forceinline__ v8f wmma16(v16h a, v16h b, v8f c) {
  // (neg_a, A, neg_b, B, c_mod, C, reuse_a, reuse_b)
  return __builtin_amdgcn_wmma_f32_16x16x32_f16(false, a, false, b, (short)0, c,
                                                false, false);
}

// A-fragment (16x32 f16): lane m = l16; element e -> k = kbase+halfL*16+e.
// Caller passes kofs = kbase + halfL*8.
__device__ __forceinline__ v16h load_a16(const _Float16* base, int ld, int l16,
                                         int kofs) {
  const _Float16* p = base + (size_t)l16 * ld + kofs + (kofs & 8);
  v8h lo = *(const v8h*)(p);
  v8h hi = *(const v8h*)(p + 8);
  v16h r;
#pragma unroll
  for (int i = 0; i < 8; ++i) { r[i] = lo[i]; r[i + 8] = hi[i]; }
  return r;
}

// ---------------------------------------------------------------------------
// 0) weight convert + transpose: wt[n*K + k] = (f16) w[k*N + n]
// ---------------------------------------------------------------------------
__global__ void k_transpose_f32_f16(const float* __restrict__ w,
                                    _Float16* __restrict__ wt, int K, int N) {
  size_t idx = (size_t)blockIdx.x * 256 + threadIdx.x;
  if (idx >= (size_t)K * N) return;
  int n = (int)(idx / K);
  int k = (int)(idx % K);
  wt[idx] = (_Float16)w[(size_t)k * N + n];
}

// ---------------------------------------------------------------------------
// 1) RMS norm: xh = x / ||x|| * sqrt(1024) * (gamma+1), f16 out
// ---------------------------------------------------------------------------
__global__ void k_rmsnorm(const float* __restrict__ x,
                          const float* __restrict__ gamma,
                          _Float16* __restrict__ xh) {
  __shared__ float red[8];
  int row = blockIdx.x;
  const float* xr = x + (size_t)row * DIM;
  float ss = 0.f;
#pragma unroll
  for (int i = threadIdx.x; i < DIM; i += 256) { float v = xr[i]; ss += v * v; }
  ss += __shfl_xor(ss, 1);  ss += __shfl_xor(ss, 2);
  ss += __shfl_xor(ss, 4);  ss += __shfl_xor(ss, 8);
  ss += __shfl_xor(ss, 16);
  if ((threadIdx.x & 31) == 0) red[threadIdx.x >> 5] = ss;
  __syncthreads();
  float tot = 0.f;
#pragma unroll
  for (int w = 0; w < 8; ++w) tot += red[w];
  float norm = fmaxf(sqrtf(tot), 1e-12f);
  float f = 32.0f / norm;   // sqrt(1024) / norm
#pragma unroll
  for (int i = threadIdx.x; i < DIM; i += 256)
    xh[(size_t)row * DIM + i] = (_Float16)(xr[i] * f * (gamma[i] + 1.0f));
}

// ---------------------------------------------------------------------------
// 2) QKV GEMM: [4096,1024]f16 x BT[3072,1024]f16 -> scatter q/k/v [b][h][n][d]
//    block: 8 waves x (32x64) = 256x64 C tile; B-frag reused for 2 wmma
// ---------------------------------------------------------------------------
__global__ void k_gemm_qkv(const _Float16* __restrict__ A,
                           const _Float16* __restrict__ BT,
                           _Float16* __restrict__ qkv) {
  int nb = blockIdx.x * 64;
  int mb = blockIdx.y * 256;
  int wave = threadIdx.x >> 5, lane = threadIdx.x & 31;
  int halfL = lane >> 4, l16 = lane & 15;
  const _Float16* Aw0 = A + (size_t)(mb + wave * 32) * DIM;
  const _Float16* Aw1 = Aw0 + (size_t)16 * DIM;

  v8f acc0[4] = {}, acc1[4] = {};
  for (int k = 0; k < DIM; k += 32) {
    v16h a0 = load_a16(Aw0, DIM, l16, k + halfL * 8);
    v16h a1 = load_a16(Aw1, DIM, l16, k + halfL * 8);
    if (k + 32 < DIM) {  // pull next K-step of streaming activations ahead
      __builtin_prefetch(Aw0 + (size_t)l16 * DIM + k + 32, 0, 3);
      __builtin_prefetch(Aw1 + (size_t)l16 * DIM + k + 32, 0, 3);
    }
#pragma unroll
    for (int t = 0; t < 4; ++t) {
      v16h b = *(const v16h*)(BT + (size_t)(nb + t * 16 + l16) * DIM + k + halfL * 16);
      acc0[t] = wmma16(a0, b, acc0[t]);
      acc1[t] = wmma16(a1, b, acc1[t]);
    }
  }
#pragma unroll
  for (int t = 0; t < 4; ++t) {
#pragma unroll
    for (int r = 0; r < 8; ++r) {
      int col = nb + t * 16 + l16;
      int which = col >> 10;            // 0:q 1:k 2:v
      int rem = col & 1023;
      int h = rem >> 6, d = rem & 63;
      float sc = (which == 0 ? 0.125f : 1.0f);  // q * dhead^-0.5
      {
        int m = mb + wave * 32 + r + 8 * halfL;
        int b = m >> 11, n = m & 2047;
        qkv[((size_t)which << 22) +
            ((((size_t)(b * HEADS + h)) * SEQ + n) * DHEAD + d)] =
            (_Float16)(acc0[t][r] * sc);
      }
      {
        int m = mb + wave * 32 + 16 + r + 8 * halfL;
        int b = m >> 11, n = m & 2047;
        qkv[((size_t)which << 22) +
            ((((size_t)(b * HEADS + h)) * SEQ + n) * DHEAD + d)] =
            (_Float16)(acc1[t][r] * sc);
      }
    }
  }
}

// ---------------------------------------------------------------------------
// 3) Flash attention. Block = 128 query rows of one (b,h); 8 waves x 16 rows.
//    Key tiles of 32; K/V staged in LDS (V transposed); online softmax.
// ---------------------------------------------------------------------------
__global__ void k_attention(const _Float16* __restrict__ qkv,
                            _Float16* __restrict__ aoh) {
  __shared__ __align__(32) _Float16 ldsK[32 * 64];    // [j][d]
  __shared__ __align__(32) _Float16 ldsVT[64 * 32];   // [d][j]
  __shared__ __align__(32) _Float16 ldsP[8][16 * 32]; // per-wave P bounce

  int nt = blockIdx.x & 15;
  int h  = (blockIdx.x >> 4) & 15;
  int b  = blockIdx.x >> 8;
  int wave = threadIdx.x >> 5, lane = threadIdx.x & 31;
  int halfL = lane >> 4, l16 = lane & 15;

  const size_t bh = (size_t)(b * HEADS + h) * SEQ * DHEAD;
  const _Float16* Qb = qkv + bh;
  const _Float16* Kb = qkv + ((size_t)1 << 22) + bh;
  const _Float16* Vb = qkv + ((size_t)2 << 22) + bh;

  int q0 = nt * 128 + wave * 16;
  v16h QA0 = load_a16(Qb + (size_t)q0 * DHEAD, DHEAD, l16, 0  + halfL * 8);
  v16h QA1 = load_a16(Qb + (size_t)q0 * DHEAD, DHEAD, l16, 32 + halfL * 8);

  v8f acc[4] = {};
  float m_i[8], l_i[8];
#pragma unroll
  for (int r = 0; r < 8; ++r) { m_i[r] = -INFINITY; l_i[r] = 0.f; }

  int trow = threadIdx.x >> 3;          // 0..31
  int tcol = (threadIdx.x & 7) * 8;     // 0..56
  _Float16* Pw = &ldsP[wave][0];

  for (int j0 = 0; j0 < SEQ; j0 += 32) {
    __syncthreads();  // previous tiles fully consumed
    {
      v8h kk = *(const v8h*)(Kb + (size_t)(j0 + trow) * DHEAD + tcol);
      *(v8h*)(ldsK + trow * 64 + tcol) = kk;
      v8h vv = *(const v8h*)(Vb + (size_t)(j0 + trow) * DHEAD + tcol);
#pragma unroll
      for (int i = 0; i < 8; ++i) ldsVT[(tcol + i) * 32 + trow] = vv[i];
    }
    __syncthreads();  // tiles ready

    // S = Q K^T for 16x32 tile (j split into two 16-wide fragments)
    v8f S0 = {}, S1 = {};
    {
      v16h b0 = *(const v16h*)(ldsK + l16 * 64 + halfL * 16);
      v16h b1 = *(const v16h*)(ldsK + (16 + l16) * 64 + halfL * 16);
      S0 = wmma16(QA0, b0, S0);
      S1 = wmma16(QA0, b1, S1);
      b0 = *(const v16h*)(ldsK + l16 * 64 + 32 + halfL * 16);
      b1 = *(const v16h*)(ldsK + (16 + l16) * 64 + 32 + halfL * 16);
      S0 = wmma16(QA1, b0, S0);
      S1 = wmma16(QA1, b1, S1);
    }

    // online softmax per row (row = r + 8*halfL; cols across lanes 0..15)
#pragma unroll
    for (int r = 0; r < 8; ++r) {
      float mx = fmaxf(S0[r], S1[r]);
      mx = fmaxf(mx, __shfl_xor(mx, 1));
      mx = fmaxf(mx, __shfl_xor(mx, 2));
      mx = fmaxf(mx, __shfl_xor(mx, 4));
      mx = fmaxf(mx, __shfl_xor(mx, 8));
      float mnew = fmaxf(m_i[r], mx);
      float corr = __expf(m_i[r] - mnew);
      m_i[r] = mnew;
      float p0 = __expf(S0[r] - mnew);
      float p1 = __expf(S1[r] - mnew);
      float s = p0 + p1;
      s += __shfl_xor(s, 1); s += __shfl_xor(s, 2);
      s += __shfl_xor(s, 4); s += __shfl_xor(s, 8);
      l_i[r] = l_i[r] * corr + s;
#pragma unroll
      for (int t = 0; t < 4; ++t) acc[t][r] = acc[t][r] * corr;
      int mrow = r + 8 * halfL;
      Pw[mrow * 32 + l16]      = (_Float16)p0;
      Pw[mrow * 32 + 16 + l16] = (_Float16)p1;
    }
    __syncthreads();  // P bounce visible (and orders LDS for the compiler)

    // out += P @ V  (K=32 over j, 4 d-fragments)
    v16h pa = load_a16(Pw, 32, l16, halfL * 8);
#pragma unroll
    for (int t = 0; t < 4; ++t) {
      v16h vb = *(const v16h*)(ldsVT + (t * 16 + l16) * 32 + halfL * 16);
      acc[t] = wmma16(pa, vb, acc[t]);
    }
  }

  // epilogue: normalize and write [b*2048+n][h*64+d] f16
#pragma unroll
  for (int r = 0; r < 8; ++r) {
    int n = q0 + r + 8 * halfL;
    float inv = 1.0f / l_i[r];
    size_t rowbase = ((size_t)(b * SEQ + n)) * DIM + h * DHEAD;
#pragma unroll
    for (int t = 0; t < 4; ++t)
      aoh[rowbase + t * 16 + l16] = (_Float16)(acc[t][r] * inv);
  }
}

// ---------------------------------------------------------------------------
// 4) Output GEMM: [4096,1024]f16 x woutT[1024,1024]f16 -> fp32 d_out
//    block: 8 waves x (32x64) = 256x64 C tile
// ---------------------------------------------------------------------------
__global__ void k_gemm_out(const _Float16* __restrict__ A,
                           const _Float16* __restrict__ BT,
                           float* __restrict__ out) {
  int nb = blockIdx.x * 64;
  int mb = blockIdx.y * 256;
  int wave = threadIdx.x >> 5, lane = threadIdx.x & 31;
  int halfL = lane >> 4, l16 = lane & 15;
  const _Float16* Aw0 = A + (size_t)(mb + wave * 32) * DIM;
  const _Float16* Aw1 = Aw0 + (size_t)16 * DIM;

  v8f acc0[4] = {}, acc1[4] = {};
  for (int k = 0; k < DIM; k += 32) {
    v16h a0 = load_a16(Aw0, DIM, l16, k + halfL * 8);
    v16h a1 = load_a16(Aw1, DIM, l16, k + halfL * 8);
    if (k + 32 < DIM) {
      __builtin_prefetch(Aw0 + (size_t)l16 * DIM + k + 32, 0, 3);
      __builtin_prefetch(Aw1 + (size_t)l16 * DIM + k + 32, 0, 3);
    }
#pragma unroll
    for (int t = 0; t < 4; ++t) {
      v16h b = *(const v16h*)(BT + (size_t)(nb + t * 16 + l16) * DIM + k + halfL * 16);
      acc0[t] = wmma16(a0, b, acc0[t]);
      acc1[t] = wmma16(a1, b, acc1[t]);
    }
  }
#pragma unroll
  for (int t = 0; t < 4; ++t) {
#pragma unroll
    for (int r = 0; r < 8; ++r) {
      int col = nb + t * 16 + l16;
      int m0 = mb + wave * 32 + r + 8 * halfL;
      out[(size_t)m0 * DIM + col] = acc0[t][r];
      int m1 = m0 + 16;
      out[(size_t)m1 * DIM + col] = acc1[t][r];
    }
  }
}

// ---------------------------------------------------------------------------
extern "C" void kernel_launch(void* const* d_in, const int* in_sizes, int n_in,
                              void* d_out, int out_size, void* d_ws, size_t ws_size,
                              hipStream_t stream) {
  const float* x     = (const float*)d_in[0];
  const float* gamma = (const float*)d_in[1];
  const float* w_qkv = (const float*)d_in[2];  // [1024, 3072]
  const float* w_out = (const float*)d_in[3];  // [1024, 1024]
  float* out = (float*)d_out;                  // [2,2048,1024]

  _Float16* ws = (_Float16*)d_ws;
  _Float16* xh    = ws;                               // 4096*1024
  _Float16* qkvh  = xh + (size_t)ROWS * DIM;          // 3 * 4096*1024
  _Float16* aoh   = qkvh + (size_t)3 * ROWS * DIM;    // 4096*1024
  _Float16* wqkvT = aoh + (size_t)ROWS * DIM;         // 3072*1024
  _Float16* woutT = wqkvT + (size_t)QKV_COLS * DIM;   // 1024*1024

  k_transpose_f32_f16<<<(DIM * QKV_COLS) / 256, 256, 0, stream>>>(w_qkv, wqkvT, DIM, QKV_COLS);
  k_transpose_f32_f16<<<(DIM * DIM) / 256, 256, 0, stream>>>(w_out, woutT, DIM, DIM);
  k_rmsnorm<<<ROWS, 256, 0, stream>>>(x, gamma, xh);
  k_gemm_qkv<<<dim3(QKV_COLS / 64, ROWS / 256), 256, 0, stream>>>(xh, wqkvT, qkvh);
  k_attention<<<BATCH * HEADS * (SEQ / 128), 256, 0, stream>>>(qkvh, aoh);
  k_gemm_out<<<dim3(DIM / 64, ROWS / 256), 256, 0, stream>>>(aoh, woutT, out);
}
